// DRNLayer_89412629168743
// MI455X (gfx1250) — matrix operand order repaired and checked by reading, not compiled, not run on hardware
//
#include <hip/hip_runtime.h>

typedef __attribute__((ext_vector_type(16))) _Float16 v16h;
typedef __attribute__((ext_vector_type(8)))  float    v8f;
typedef _Float16 half_t;

#define N_LOW 128
#define N_UP  128
#define Q_LOW 64
#define Q_UP  64

#define LOG2E 1.4426950408889634f

// Block: j = blockIdx.x (n_upper), blockIdx.y = batch half (32 batches each).
// 256 threads = 8 wave32s. Wave w handles output tile (lt = w&3, it = w>>2)
// of the 64(l) x 32(i) Pw block for each k.
__global__ __launch_bounds__(256)
void drn_layer_kernel(const float* __restrict__ P,
                      const float* __restrict__ weight,
                      const float* __restrict__ bias_abs,
                      const float* __restrict__ bias_q,
                      const float* __restrict__ lambda_abs,
                      const float* __restrict__ lambda_q,
                      float* __restrict__ out)
{
    __shared__ __align__(32) half_t eT[128];          // Toeplitz exp(w*logD) values, d=0..126
    __shared__ __align__(32) half_t Pl[32][64];       // P[:,k,:] f16 tile, i-major
    __shared__ float  S[32][Q_UP + 1];                // log2-domain rows (stride 65: no bank conflicts)

    const int tid   = threadIdx.x;
    const int lane  = tid & 31;
    const int wave  = tid >> 5;
    const int j     = blockIdx.x;
    const int iBase = blockIdx.y * 32;

    const int lt   = wave & 3;            // l-tile index (0..3)
    const int it   = wave >> 2;           // i-tile index (0..1)
    const int hk   = (lane >> 4) * 8;     // ISA-layout K offset per lane-half (unused)
    const int hk16 = (lane >> 4) * 16;    // B-frag K offset per lane-half
    const int lnM  = lane & 15;
    const int lA   = lt * 16 + lnM;       // A-fragment row (l)
    const int cM0  = (lane >> 4) * 8;     // C-fragment M base

    // Toeplitz generator pre-scaled by log2(e): exp(w*g) = exp2(w * g2)
    float gval2 = 0.f;
    if (tid < 127) {
        float fd = (float)(tid - 63) * (1.0f / 64.0f);
        gval2 = -fd * fd * LOG2E;
    }

    float acc[8], prod[8];                // acc accumulates log2(prod)
    #pragma unroll
    for (int r = 0; r < 8; ++r) { acc[r] = 0.f; prod[r] = 1.f; }

    for (int k = 0; k < N_LOW; ++k) {
        float w = weight[j * N_LOW + k];          // uniform -> scalar load

        __syncthreads();                          // protect LDS vs previous iter readers
        if (tid < 127)                            // 127 exps instead of 4096 (Toeplitz)
            eT[tid] = (half_t)__builtin_amdgcn_exp2f(w * gval2);

        // Stage P[iBase..iBase+31, k, :] -> f16 LDS (8 threads/row, 8 cols each)
        {
            int r = tid >> 3;
            int c = (tid & 7) * 8;
            const float* src = P + ((size_t)(iBase + r) * N_LOW + k) * Q_LOW + c;
            __builtin_prefetch(src + Q_LOW, 0, 0);   // next-k row -> global_prefetch_b8
            float4 f0 = *(const float4*)(src);
            float4 f1 = *(const float4*)(src + 4);
            half_t* dst = &Pl[r][c];
            dst[0] = (half_t)f0.x; dst[1] = (half_t)f0.y;
            dst[2] = (half_t)f0.z; dst[3] = (half_t)f0.w;
            dst[4] = (half_t)f1.x; dst[5] = (half_t)f1.y;
            dst[6] = (half_t)f1.z; dst[7] = (half_t)f1.w;
        }
        __syncthreads();

        // A fragments (contiguous K, same order as B): element p -> K = hk16 + p
        v16h a0, a1;
        #pragma unroll
        for (int p = 0; p < 16; ++p) {
            int Kp = hk16 + p;
            a0[p] = eT[Kp      - lA + 63];    // T[l][m] = e[m - l + 63], m_base = 0
            a1[p] = eT[Kp + 32 - lA + 63];    // m_base = 32
        }
        // B fragments: K = hk16 + p -> 16 contiguous halfs in the i-major row
        int iRow = it * 16 + lnM;
        v16h b0 = *(const v16h*)&Pl[iRow][hk16];
        v16h b1 = *(const v16h*)&Pl[iRow][32 + hk16];

        // Pw tile = T[16l x 64m] x P^T[64m x 16i], K=64 via two chained WMMAs
        v8f c = {};
        c = __builtin_amdgcn_wmma_f32_16x16x32_f16(false, a0, false, b0,
                                                   (short)0, c, false, false);
        c = __builtin_amdgcn_wmma_f32_16x16x32_f16(false, a1, false, b1,
                                                   (short)0, c, false, false);

        // clip then fold into running product; one raw v_log_f32 per 8 k's
        // (range-safe: realistic Pw in [~5,100] -> prod <= ~1e16 << f32 max;
        //  clip >= 1e-15 keeps everything normal, so no denorm guards needed)
        #pragma unroll
        for (int r = 0; r < 8; ++r) {
            float v = fminf(fmaxf(c[r], 1e-15f), 1e15f);
            prod[r] *= v;
        }
        if ((k & 7) == 7) {
            #pragma unroll
            for (int r = 0; r < 8; ++r) {
                acc[r] += __builtin_amdgcn_logf(prod[r]);   // log2
                prod[r] = 1.f;
            }
        }
    }

    // Bias pre-scaled by log2(e): softmax_e(x) == softmax_2(x*log2e), and
    // acc already holds logsum*log2e (it is a sum of log2's).
    float bq = bias_q[j] * LOG2E, ba = bias_abs[j] * LOG2E;
    float lq = lambda_q[j], la = lambda_abs[j];
    __syncthreads();
    #pragma unroll
    for (int r = 0; r < 8; ++r) {
        int l = lt * 16 + cM0 + r;
        int i = it * 16 + lnM;
        float s  = (float)l * (1.0f / 64.0f);
        float dq = s - lq;
        float bias = -bq * dq * dq - ba * fabsf(s - la);
        S[i][l] = acc[r] + bias;
    }
    __syncthreads();

    // Softmax over l per batch row (32 rows, one thread each), base-2 domain
    if (tid < 32) {
        int i = tid;
        float m = -3.4e38f;
        #pragma unroll 4
        for (int l = 0; l < Q_UP; ++l) m = fmaxf(m, S[i][l]);
        float sum = 0.f;
        #pragma unroll 4
        for (int l = 0; l < Q_UP; ++l) {
            float e = __builtin_amdgcn_exp2f(S[i][l] - m);
            S[i][l] = e;
            sum += e;
        }
        float inv = 1.0f / sum;
        float* dst = out + ((size_t)(iBase + i) * N_UP + j) * Q_UP;
        #pragma unroll 4
        for (int l = 0; l < Q_UP; ++l) dst[l] = S[i][l] * inv;
    }
}

extern "C" void kernel_launch(void* const* d_in, const int* in_sizes, int n_in,
                              void* d_out, int out_size, void* d_ws, size_t ws_size,
                              hipStream_t stream) {
    (void)in_sizes; (void)n_in; (void)out_size; (void)d_ws; (void)ws_size;
    const float* P          = (const float*)d_in[0];
    const float* weight     = (const float*)d_in[1];
    const float* bias_abs   = (const float*)d_in[2];
    const float* bias_q     = (const float*)d_in[3];
    const float* lambda_abs = (const float*)d_in[4];
    const float* lambda_q   = (const float*)d_in[5];
    float* out = (float*)d_out;

    dim3 grid(N_UP, 2);    // 128 j-columns x 2 batch halves = 256 workgroups
    dim3 block(256);       // 8 wave32s
    drn_layer_kernel<<<grid, block, 0, stream>>>(P, weight, bias_abs, bias_q,
                                                 lambda_abs, lambda_q, out);
}
